// ModulatedConv2d_88776974008852
// MI455X (gfx1250) — matrix operand order repaired and checked by reading, not compiled, not run on hardware
//
#include <hip/hip_runtime.h>

// ---------------------------------------------------------------------------
// Modulated Conv2d (StyleGAN2) for gfx1250:
//   b=8, c_in=c_out=512, k=3, h=w=64.  Implicit GEMM over 9 taps with
//   v_wmma_f32_16x16x32_bf16 (f32 accumulate), double-buffered LDS fed by
//   GLOBAL_LOAD_ASYNC_TO_LDS_B128 (ASYNCcnt).  64-channel K-stages to
//   amortize barriers (16 WMMA / wave / stage).
// ---------------------------------------------------------------------------

typedef __attribute__((ext_vector_type(16))) __bf16        v16bf;
typedef __attribute__((ext_vector_type(8)))  float         v8f;
typedef __attribute__((ext_vector_type(4)))  unsigned int  u32x4;
typedef __attribute__((ext_vector_type(4)))  int           i32x4;

#define NB    8
#define CIN   512
#define COUT  512
#define NSF   512
#define HH    64
#define WW    64
#define HW    4096
#define KTAPS 9
#define XH    66            // haloed spatial extent (1-pixel zero border)
#define NST   (KTAPS * 8)   // 72 K-stages of 64 channels
#define LROW  72            // LDS row stride in bf16 elems (144B, 16B-aligned)

#ifndef __has_builtin
#define __has_builtin(x) 0
#endif

#if __has_builtin(__builtin_amdgcn_global_load_async_to_lds_b128) && \
    __has_builtin(__builtin_amdgcn_s_wait_asynccnt)
#define USE_ASYNC 1
#else
#define USE_ASYNC 0
#endif

#if USE_ASYNC
// Builtin signature (probe-verified): param0 = V4i AS(1) (global src),
// param1 = V4i AS(3) (LDS dst), then two immediate ints (offset, cpol).
typedef __attribute__((address_space(1))) i32x4 g_i32x4;
typedef __attribute__((address_space(3))) i32x4 l_i32x4;
static __device__ __forceinline__ void async_b128(const void* g, void* l) {
  __builtin_amdgcn_global_load_async_to_lds_b128(
      (g_i32x4*)(unsigned long long)g,
      (l_i32x4*)(unsigned int)(unsigned long long)l, 0, 0);
}
#endif

static __device__ __forceinline__ unsigned short f2bf(float f) {
  unsigned int u = __float_as_uint(f);
  u += 0x7FFFu + ((u >> 16) & 1u);
  return (unsigned short)(u >> 16);
}

// ---------------- prep kernels (tiny) ----------------

// s[b][ci] = mod_b[ci] + sum_f style[b][f] * mod_w[ci][f]
__global__ void k_modulate(const float* __restrict__ style,
                           const float* __restrict__ mod_w,
                           const float* __restrict__ mod_b,
                           float* __restrict__ s) {
  int i = blockIdx.x * blockDim.x + threadIdx.x;   // 4096
  int b = i >> 9, ci = i & 511;
  const float* st = style + (size_t)b * NSF;
  const float* mw = mod_w + (size_t)ci * NSF;
  float acc = mod_b[ci];
  for (int f = 0; f < NSF; ++f) acc += st[f] * mw[f];
  s[i] = acc;
}

// wsq[co][ci] = sum_tap weight[co][ci][tap]^2
__global__ void k_wsq(const float* __restrict__ w, float* __restrict__ wsq) {
  int i = blockIdx.x * blockDim.x + threadIdx.x;   // 262144
  const float* p = w + (size_t)i * KTAPS;
  float acc = 0.f;
#pragma unroll
  for (int t = 0; t < KTAPS; ++t) { float v = p[t]; acc += v * v; }
  wsq[i] = acc;
}

// demod[b][co] = rsqrt( sum_ci wsq[co][ci]*s[b][ci]^2 + 1e-8 )
__global__ void k_demod(const float* __restrict__ wsq,
                        const float* __restrict__ s,
                        float* __restrict__ demod) {
  int i = blockIdx.x * blockDim.x + threadIdx.x;   // 4096
  int b = i >> 9, co = i & 511;
  const float* q  = wsq + (size_t)co * CIN;
  const float* sb = s   + (size_t)b  * CIN;
  float acc = 0.f;
  for (int ci = 0; ci < CIN; ++ci) { float sv = sb[ci]; acc += q[ci] * sv * sv; }
  demod[i] = rsqrtf(acc + 1e-8f);
}

// wmodb[b][tap][co][ci] = bf16( w[co][ci][tap] * s[b][ci] * demod[b][co] )
__global__ void k_wmod(const float* __restrict__ w,
                       const float* __restrict__ s,
                       const float* __restrict__ demod,
                       unsigned short* __restrict__ wmodb) {
  int i  = blockIdx.x * blockDim.x + threadIdx.x;  // 18,874,368
  int ci = i & 511;
  int co = (i >> 9) & 511;
  int q  = i >> 18;                                // b*9 + tap
  int tap = q % KTAPS, b = q / KTAPS;
  float v = w[(size_t)(co * CIN + ci) * KTAPS + tap]
          * s[b * CIN + ci] * demod[b * COUT + co];
  wmodb[i] = f2bf(v);
}

// zero the haloed activation buffer (u32x4 stores)
__global__ void k_zero(u32x4* __restrict__ p) {
  size_t i = (size_t)blockIdx.x * blockDim.x + threadIdx.x;  // 2,230,272
  p[i] = (u32x4){0u, 0u, 0u, 0u};
}

// ---------------- x fp32 [b][c][h][w] -> bf16 xt[b][y+1][x+1][c] ------------
// LDS-staged transpose: block handles 32 channels x 128 pixels.
__global__ void k_xconv(const float* __restrict__ x,
                        unsigned short* __restrict__ xt) {
  __shared__ unsigned short lds[128 * 40];   // padded row stride (80B, 16B-aligned)
  int blk = blockIdx.x;                      // 8 * 16 * 32 = 4096
  int b  = blk >> 9;
  int c0 = ((blk >> 5) & 15) * 32;
  int pt = (blk & 31) * 128;
  int t  = threadIdx.x;

  int ch = t & 31;
  int pg = (t >> 5) * 16;
  const float* src = x + ((size_t)(b * CIN + c0 + ch) * HW + pt + pg);
#pragma unroll
  for (int j = 0; j < 16; ++j)
    lds[(pg + j) * 40 + ch] = f2bf(src[j]);
  __syncthreads();

  int pix = t >> 1, half = t & 1;
  const u32x4* lp = (const u32x4*)&lds[pix * 40 + half * 16];
  u32x4 q0 = lp[0], q1 = lp[1];
  int gp = pt + pix;
  int yy = gp >> 6, xx = gp & 63;
  u32x4* dst = (u32x4*)(xt + (size_t)((b * XH + yy + 1) * XH + xx + 1) * CIN
                           + c0 + half * 16);
  dst[0] = q0; dst[1] = q1;
}

// ---------------- main conv: implicit GEMM with WMMA bf16 ----------------
// Block: 128 c_out x 128 pixels; 8 waves in 2(M) x 4(N) grid; each wave 4x2
// 16x16 accumulators.  72 stages of K=64, double-buffered async-fed LDS.
__global__ __launch_bounds__(256)
void k_conv(const unsigned short* __restrict__ xt,     // [b][y+1][x+1][ci] bf16 (haloed)
            const unsigned short* __restrict__ wmodb,  // [b][tap][co][ci]  bf16
            float* __restrict__ out) {                 // [b][co][pix] f32
  __shared__ unsigned short sm[4][128 * LROW];  // {bufA0,bufB0,bufA1,bufB1} = 72KB

  int blk = blockIdx.x;                  // 8 * 4 * 32 = 1024
  int pt  = (blk & 31) * 128;
  int co0 = ((blk >> 5) & 3) * 128;
  int b   = blk >> 7;

  int t    = threadIdx.x;
  int lane = t & 31;
  int wv   = t >> 5;
  int wm_  = wv >> 2;                    // 0..1 : M sub-tile
  int wn_  = wv & 3;                     // 0..3 : N sub-tile

  int row  = t >> 1;                     // 0..127 loader row
  int half = t & 1;                      // 64B half of a 128B row

  int gp = pt + row;
  int py = gp >> 6, px = gp & 63;

  v8f acc[4][2];
#pragma unroll
  for (int mt = 0; mt < 4; ++mt)
#pragma unroll
    for (int nt = 0; nt < 2; ++nt)
      acc[mt][nt] = (v8f){0.f, 0.f, 0.f, 0.f, 0.f, 0.f, 0.f, 0.f};

  const int l15   = lane & 15;
  const int akofs = (lane >> 4) * 8;     // A: K elems {0..7,16..23} / {8..15,24..31}
  const int bkofs = (lane >> 4) * 8;     // B: K elems {0..7,16..23} / {8..15,24..31}
  const int ldofs = row * LROW + half * 32;   // loader LDS offset (elems)

#if !USE_ASYNC
  u32x4 ra[4], rb[4];                    // fallback staging registers
#endif

  // ---- addresses for stage `it` (tap = it/8, ci0 = (it%8)*64) ----
  auto tile_addrs = [&](int it, const unsigned short*& asrc, const unsigned short*& bsrc) {
    int tap = it >> 3;
    int ci0 = (it & 7) << 6;
    int ky = tap / 3 - 1, kx = tap - (tap / 3) * 3 - 1;
    int y = py + ky + 1, xx = px + kx + 1;          // haloed coords, always valid
    asrc = wmodb + (size_t)((b * KTAPS + tap) * COUT + co0 + row) * CIN
                 + ci0 + half * 32;
    bsrc = xt + (size_t)((b * XH + y) * XH + xx) * CIN + ci0 + half * 32;
  };

#if USE_ASYNC
  auto issue = [&](int it, int bsel) {
    const unsigned short *asrc, *bsrc;
    tile_addrs(it, asrc, bsrc);
    unsigned short* la = &sm[bsel * 2][ldofs];
    unsigned short* lb = &sm[bsel * 2 + 1][ldofs];
#pragma unroll
    for (int j = 0; j < 4; ++j) {
      async_b128(asrc + j * 8, la + j * 8);
      async_b128(bsrc + j * 8, lb + j * 8);
    }
  };

  issue(0, 0);
  int buf = 0;
  for (int it = 0; it < NST; ++it) {
    if (it + 1 < NST) {
      issue(it + 1, buf ^ 1);
      __builtin_amdgcn_s_wait_asynccnt(8);   // stage `it` landed (mine)
    } else {
      __builtin_amdgcn_s_wait_asynccnt(0);
    }
    __syncthreads();                         // everyone's stage `it` landed
#else
  // fallback: register-staged double buffering
  {
    const unsigned short *asrc, *bsrc;
    tile_addrs(0, asrc, bsrc);
#pragma unroll
    for (int j = 0; j < 4; ++j) {
      ra[j] = ((const u32x4*)asrc)[j];
      rb[j] = ((const u32x4*)bsrc)[j];
      *(u32x4*)&sm[0][ldofs + j * 8] = ra[j];
      *(u32x4*)&sm[1][ldofs + j * 8] = rb[j];
    }
  }
  __syncthreads();
  int buf = 0;
  for (int it = 0; it < NST; ++it) {
    if (it + 1 < NST) {
      const unsigned short *asrc, *bsrc;
      tile_addrs(it + 1, asrc, bsrc);
#pragma unroll
      for (int j = 0; j < 4; ++j) {
        ra[j] = ((const u32x4*)asrc)[j];
        rb[j] = ((const u32x4*)bsrc)[j];
      }
    }
#endif

    // ---- compute on lds buffer `buf`: two K=32 chunks ----
    {
      const unsigned short* lA = sm[buf * 2];
      const unsigned short* lB = sm[buf * 2 + 1];
      union Frag { u32x4 q[2]; v16bf v; };
#pragma unroll
      for (int kc = 0; kc < 2; ++kc) {
        const int kbase = kc * 32;
        Frag af[4], bfr[2];
#pragma unroll
        for (int nt = 0; nt < 2; ++nt) {
          int r = wn_ * 32 + nt * 16 + l15;
          bfr[nt].q[0] = *(const u32x4*)&lB[r * LROW + kbase + bkofs];
          bfr[nt].q[1] = *(const u32x4*)&lB[r * LROW + kbase + bkofs + 16];
        }
#pragma unroll
        for (int mt = 0; mt < 4; ++mt) {
          int r = wm_ * 64 + mt * 16 + l15;
          af[mt].q[0] = *(const u32x4*)&lA[r * LROW + kbase + akofs];
          af[mt].q[1] = *(const u32x4*)&lA[r * LROW + kbase + akofs + 16];
        }
#pragma unroll
        for (int mt = 0; mt < 4; ++mt)
#pragma unroll
          for (int nt = 0; nt < 2; ++nt)
            acc[mt][nt] = __builtin_amdgcn_wmma_f32_16x16x32_bf16(
                false, af[mt].v, false, bfr[nt].v,
                (short)0, acc[mt][nt], false, false);
      }
    }

    __syncthreads();                         // reads of `buf` done -> reusable
#if !USE_ASYNC
    if (it + 1 < NST) {
      int nb = buf ^ 1;
#pragma unroll
      for (int j = 0; j < 4; ++j) {
        *(u32x4*)&sm[nb * 2][ldofs + j * 8]     = ra[j];
        *(u32x4*)&sm[nb * 2 + 1][ldofs + j * 8] = rb[j];
      }
    }
    __syncthreads();
#endif
    buf ^= 1;
  }

  // store: D 16x16 f32 layout -> VGPR r, lanes 0-15: M=r, lanes 16-31: M=r+8
  int lh = lane >> 4;
#pragma unroll
  for (int mt = 0; mt < 4; ++mt) {
#pragma unroll
    for (int nt = 0; nt < 2; ++nt) {
      int co  = co0 + wm_ * 64 + mt * 16 + lh * 8;
      int pix = pt + wn_ * 32 + nt * 16 + l15;
      float* o = out + (size_t)b * COUT * HW + (size_t)co * HW + pix;
#pragma unroll
      for (int r = 0; r < 8; ++r) o[(size_t)r * HW] = acc[mt][nt][r];
    }
  }
}

// ---------------- host launcher ----------------
extern "C" void kernel_launch(void* const* d_in, const int* in_sizes, int n_in,
                              void* d_out, int out_size, void* d_ws, size_t ws_size,
                              hipStream_t stream) {
  (void)in_sizes; (void)n_in; (void)out_size; (void)ws_size;
  const float* x      = (const float*)d_in[0];   // (8,512,64,64)
  const float* style  = (const float*)d_in[1];   // (8,512)
  const float* mod_w  = (const float*)d_in[2];   // (512,512)
  const float* mod_b  = (const float*)d_in[3];   // (512,)
  const float* weight = (const float*)d_in[4];   // (1,512,512,3,3)
  float* out = (float*)d_out;                    // (8,512,64,64) f32

  // workspace layout (needs ~76 MB):
  //   [0,      16KB)   s      (8x512 f32)
  //   [16KB,   32KB)   demod  (8x512 f32)
  //   [32KB,  1.03MB)  wsq    (512x512 f32)
  //   [2MB,   ~37.7MB) xt     bf16 [b][66][66][ci]  (zero halo)
  //   [40MB,  76MB)    wmodb  bf16 [b][tap][co][ci]
  char* ws = (char*)d_ws;
  float* s     = (float*)ws;
  float* demod = s + 4096;
  float* wsq   = demod + 4096;
  unsigned short* xt    = (unsigned short*)(ws + (2ull  << 20));
  unsigned short* wmodb = (unsigned short*)(ws + (40ull << 20));

  k_modulate<<<16,    256, 0, stream>>>(style, mod_w, mod_b, s);
  k_wsq     <<<1024,  256, 0, stream>>>(weight, wsq);
  k_demod   <<<16,    256, 0, stream>>>(wsq, s, demod);
  k_wmod    <<<73728, 256, 0, stream>>>(weight, s, demod, wmodb);
  k_zero    <<<8712,  256, 0, stream>>>((u32x4*)xt);   // 8*66*66*512 elems / 8
  k_xconv   <<<4096,  256, 0, stream>>>(x, xt);
  k_conv    <<<1024,  256, 0, stream>>>(xt, wmodb, out);
}
